// gruNet_80960133529704
// MI455X (gfx1250) — matrix operand order repaired and checked by reading, not compiled, not run on hardware
//
#include <hip/hip_runtime.h>

// ---------------------------------------------------------------------------
// Problem dimensions (match reference)
// ---------------------------------------------------------------------------
#define B_      1024
#define T_      2048
#define IN_     32
#define H_      7
#define G_      21        // 3*H
#define OUT_    8
#define D1_     1024
#define D2_     2048
#define OSZ_    14336     // H*T

typedef __bf16 bf16_t;
typedef __attribute__((ext_vector_type(16))) __bf16 v16bf;
typedef __attribute__((ext_vector_type(8)))  float  v8f;
typedef __attribute__((ext_vector_type(2)))  float  v2f;

// ---------------------------------------------------------------------------
// Kernel 1: gi[bt, g] = x[bt, :] @ w_ih[g, :]^T + b_ih[g]   (f32 WMMA, K=32)
// One wave per 16-row M tile of (B*T) x 21. w_ih fragments preloaded.
// A frag (16x4 f32): lane m=lane&15, k = kbase + (lane>>4)*2 + {0,1}
// B frag (4x16 f32): lane n=lane&15, same k pattern (B[k][n] = w_ih[n][k])
// C (16x16 f32): VGPR r, lane c -> (M = r + 8*(lane>>4), N = lane&15)
// ---------------------------------------------------------------------------
__device__ __forceinline__ v2f ld_v2f(const float* p) {
  union { float2 f; v2f v; } u;
  u.f = *(const float2*)p;
  return u.v;
}

__global__ void __launch_bounds__(256)
k_gi(const float* __restrict__ x, const float* __restrict__ w_ih,
     const float* __restrict__ b_ih, float* __restrict__ gi)
{
  const int lane = threadIdx.x & 31;
  const int wid  = (int)((blockIdx.x * blockDim.x + threadIdx.x) >> 5);
  const long row0 = (long)wid * 16;
  const int mlo   = lane & 15;
  const int kpair = (lane >> 4) << 1;   // 0 or 2

  // Preload B fragments: 8 K-steps x 2 N-tiles (gates 0..15, 16..20 padded)
  v2f bfrag[8][2];
  const int n1 = 16 + mlo;
#pragma unroll
  for (int kk = 0; kk < 8; ++kk) {
    const int k = kk * 4 + kpair;
    bfrag[kk][0] = ld_v2f(w_ih + mlo * IN_ + k);
    if (n1 < G_) bfrag[kk][1] = ld_v2f(w_ih + n1 * IN_ + k);
    else { v2f z; z[0] = 0.f; z[1] = 0.f; bfrag[kk][1] = z; }
  }

  v8f acc0 = {};
  v8f acc1 = {};
  const float* xrow = x + (row0 + mlo) * IN_;
#pragma unroll
  for (int kk = 0; kk < 8; ++kk) {
    const int k = kk * 4 + kpair;
    v2f a = ld_v2f(xrow + k);
    acc0 = __builtin_amdgcn_wmma_f32_16x16x4_f32(false, a, false, bfrag[kk][0],
                                                 (short)0, acc0, false, false);
    acc1 = __builtin_amdgcn_wmma_f32_16x16x4_f32(false, a, false, bfrag[kk][1],
                                                 (short)0, acc1, false, false);
  }

  const int hi16 = lane >> 4;
#pragma unroll
  for (int r = 0; r < 8; ++r) {
    const long m = row0 + r + 8 * hi16;
    gi[m * G_ + mlo] = acc0[r] + b_ih[mlo];
    if (n1 < G_) gi[m * G_ + n1] = acc1[r] + b_ih[n1];
  }
}

// ---------------------------------------------------------------------------
// Kernel 2: f32 -> bf16 conversion (grid-stride)
// ---------------------------------------------------------------------------
__global__ void k_cvt(const float* __restrict__ in, bf16_t* __restrict__ out, long n)
{
  long i = (long)blockIdx.x * blockDim.x + threadIdx.x;
  const long stride = (long)gridDim.x * blockDim.x;
  for (; i < n; i += stride) out[i] = (bf16_t)in[i];
}

// ---------------------------------------------------------------------------
// Kernel 3: GRU scan. One wave per batch element; lane l<21 owns gate row l:
//   rows 0..6 = r, 7..13 = z, 14..20 = n. h replicated across lanes.
// Output h_t stored as bf16 directly into flat[b, t*7+j] (feeds bf16 GEMM1).
// ---------------------------------------------------------------------------
__global__ void __launch_bounds__(128)
k_gru(const float* __restrict__ gi, const float* __restrict__ w_hh,
      const float* __restrict__ b_hh, bf16_t* __restrict__ flat)
{
  const int lane = threadIdx.x & 31;
  const int b    = blockIdx.x * 4 + (threadIdx.x >> 5);
  const int l    = (lane < G_) ? lane : 0;

  float wr[H_];
#pragma unroll
  for (int j = 0; j < H_; ++j) wr[j] = w_hh[l * H_ + j];
  const float bh = b_hh[l];

  float h[H_];
#pragma unroll
  for (int j = 0; j < H_; ++j) h[j] = 0.f;

  const float* g = gi + (long)b * T_ * G_;
  bf16_t* outp = flat + (long)b * OSZ_;

  for (int t = 0; t < T_; ++t) {
    const float gv = g[l];                        // 21 contiguous f32 per wave
    float racc = bh;
#pragma unroll
    for (int j = 0; j < H_; ++j) racc = fmaf(h[j], wr[j], racc);

    const float s = 1.f / (1.f + __expf(-(gv + racc)));   // sigmoid (r/z lanes)
    const float r = __shfl(s, (lane >= 14) ? lane - 14 : lane, 32);
    const float nv = tanhf(fmaf(r, racc, gv));            // n lanes
    const float z = __shfl(s, (lane >= 7) ? lane - 7 : lane, 32);

    float hold = 0.f;
#pragma unroll
    for (int j = 0; j < H_; ++j) hold = (lane - 14 == j) ? h[j] : hold;
    const float hnew = (1.f - z) * nv + z * hold;

    if (lane >= 14 && lane < 21) outp[t * H_ + (lane - 14)] = (bf16_t)hnew;
#pragma unroll
    for (int j = 0; j < H_; ++j) h[j] = __shfl(hnew, 14 + j, 32);
    g += G_;
  }
}

// ---------------------------------------------------------------------------
// Kernel 4: C = act(A(MxK,bf16) * B(NxK,bf16)^T + bias), bf16 WMMA 16x16x32.
// One wave computes a 64x64 block = 4x4 register-blocked 16x16 tiles.
// A frag: lane m=lane&15, K elems at kb=(lane>>4)*8: [kb..kb+7] and [kb+16..kb+23]
// B frag: lane n=lane&15, same K as A at kb: [kb..kb+7] and [kb+16..kb+23]
// ---------------------------------------------------------------------------
__device__ __forceinline__ v16bf make_frag(const bf16_t* p, int off2)
{
  union { uint4 u[2]; v16bf v; } u;
  u.u[0] = *(const uint4*)p;
  u.u[1] = *(const uint4*)(p + off2);
  return u.v;
}

__global__ void __launch_bounds__(256)
k_gemm(const bf16_t* __restrict__ A, const bf16_t* __restrict__ Bm,
       const float* __restrict__ bias, float* __restrict__ outF,
       bf16_t* __restrict__ outB, int M, int N, int K, int relu)
{
  const int lane   = threadIdx.x & 31;
  const int wid    = (int)((blockIdx.x * blockDim.x + threadIdx.x) >> 5);
  const int tilesN = N >> 6;
  const int m0 = (wid / tilesN) * 64;
  const int n0 = (wid % tilesN) * 64;
  if (m0 >= M) return;

  const int rlo = lane & 15;
  const int kbA = (lane >> 4) << 3;   // 0 / 8  elements
  const int ksB = (lane >> 4) << 4;   // 0 / 16 elements

  v8f acc[4][4];
#pragma unroll
  for (int i = 0; i < 4; ++i)
#pragma unroll
    for (int j = 0; j < 4; ++j) acc[i][j] = (v8f){};

  for (int k0 = 0; k0 < K; k0 += 32) {
    v16bf af[4], bfv[4];
#pragma unroll
    for (int i = 0; i < 4; ++i)
      af[i] = make_frag(A + (size_t)(m0 + i * 16 + rlo) * K + k0 + kbA, 16);
#pragma unroll
    for (int j = 0; j < 4; ++j)
      bfv[j] = make_frag(Bm + (size_t)(n0 + j * 16 + rlo) * K + k0 + kbA, 16);
#pragma unroll
    for (int i = 0; i < 4; ++i)
#pragma unroll
      for (int j = 0; j < 4; ++j)
        acc[i][j] = __builtin_amdgcn_wmma_f32_16x16x32_bf16(
            false, af[i], false, bfv[j], (short)0, acc[i][j], false, false);
  }

  const int hi16 = lane >> 4;
#pragma unroll
  for (int i = 0; i < 4; ++i) {
#pragma unroll
    for (int j = 0; j < 4; ++j) {
#pragma unroll
      for (int r = 0; r < 8; ++r) {
        const int m = m0 + i * 16 + r + 8 * hi16;
        const int n = n0 + j * 16 + rlo;
        float v = acc[i][j][r] + bias[n];
        if (relu) v = v > 0.f ? v : 0.f;
        if (outF) outF[(size_t)m * N + n] = v;
        if (outB) outB[(size_t)m * N + n] = (bf16_t)v;
      }
    }
  }
}

// ---------------------------------------------------------------------------
// Kernel 5: out[b, :8] = h2[b, :] @ w3^T + b3   (f32, wave-per-row reduction)
// ---------------------------------------------------------------------------
__global__ void __launch_bounds__(128)
k_head(const float* __restrict__ h2, const float* __restrict__ w3,
       const float* __restrict__ b3, float* __restrict__ out)
{
  const int lane = threadIdx.x & 31;
  const int b    = blockIdx.x * 4 + (threadIdx.x >> 5);
  float acc[OUT_];
#pragma unroll
  for (int o = 0; o < OUT_; ++o) acc[o] = 0.f;

  const float* hrow = h2 + (size_t)b * D2_;
  for (int k = lane; k < D2_; k += 32) {
    const float hv = hrow[k];
#pragma unroll
    for (int o = 0; o < OUT_; ++o) acc[o] = fmaf(hv, w3[o * D2_ + k], acc[o]);
  }
#pragma unroll
  for (int off = 16; off > 0; off >>= 1)
#pragma unroll
    for (int o = 0; o < OUT_; ++o) acc[o] += __shfl_down(acc[o], off, 32);

  if (lane == 0) {
#pragma unroll
    for (int o = 0; o < OUT_; ++o) out[b * OUT_ + o] = acc[o] + b3[o];
  }
}

// ---------------------------------------------------------------------------
// Launcher
// ---------------------------------------------------------------------------
extern "C" void kernel_launch(void* const* d_in, const int* in_sizes, int n_in,
                              void* d_out, int out_size, void* d_ws, size_t ws_size,
                              hipStream_t stream)
{
  const float* x    = (const float*)d_in[0];
  const float* w_ih = (const float*)d_in[1];
  const float* w_hh = (const float*)d_in[2];
  const float* b_ih = (const float*)d_in[3];
  const float* b_hh = (const float*)d_in[4];
  const float* w1   = (const float*)d_in[5];
  const float* b1   = (const float*)d_in[6];
  const float* w2   = (const float*)d_in[7];
  const float* b2   = (const float*)d_in[8];
  const float* w3   = (const float*)d_in[9];
  const float* b3   = (const float*)d_in[10];
  float* out = (float*)d_out;

  // Workspace layout (256B aligned regions)
  char* ws = (char*)d_ws;
  auto carve = [&](size_t bytes) -> char* {
    char* p = ws;
    ws += (bytes + 255) & ~(size_t)255;
    return p;
  };
  float*  gi    = (float*)carve((size_t)B_ * T_ * G_ * sizeof(float));   // 176 MB
  bf16_t* flatB = (bf16_t*)carve((size_t)B_ * OSZ_ * sizeof(bf16_t));    //  28 MB
  bf16_t* w1B   = (bf16_t*)carve((size_t)D1_ * OSZ_ * sizeof(bf16_t));   //  28 MB
  bf16_t* w2B   = (bf16_t*)carve((size_t)D2_ * D1_ * sizeof(bf16_t));    //   4 MB
  bf16_t* h1B   = (bf16_t*)carve((size_t)B_ * D1_ * sizeof(bf16_t));     //   2 MB
  float*  h2F   = (float*)carve((size_t)B_ * D2_ * sizeof(float));       //   8 MB

  // 1) Input projection via f32 WMMA: (B*T)/16 tiles, 8 waves per 256-thr block
  k_gi<<<(B_ * T_ / 16) / 8, 256, 0, stream>>>(x, w_ih, b_ih, gi);

  // 2) Weight conversions (independent of k_gi, overlap on the stream queue)
  k_cvt<<<2048, 256, 0, stream>>>(w1, w1B, (long)D1_ * OSZ_);
  k_cvt<<<1024, 256, 0, stream>>>(w2, w2B, (long)D2_ * D1_);

  // 3) GRU scan: 1024 waves (one per batch chain)
  k_gru<<<B_ / 4, 128, 0, stream>>>(gi, w_hh, b_hh, flatB);

  // 4) h1 = relu(flat @ w1^T + b1)  [bf16 WMMA, 256 waves]
  k_gemm<<<(16 * 16) / 8, 256, 0, stream>>>(flatB, w1B, b1, nullptr, h1B,
                                            B_, D1_, OSZ_, 1);
  // 5) h2 = relu(h1 @ w2^T + b2)    [bf16 WMMA, 512 waves]
  k_gemm<<<(16 * 32) / 8, 256, 0, stream>>>(h1B, w2B, b2, h2F, nullptr,
                                            B_, D2_, D1_, 1);
  // 6) out = h2 @ w3^T + b3
  k_head<<<B_ / 4, 128, 0, stream>>>(h2F, w3, b3, out);
}
